// GCN_10642928959813
// MI455X (gfx1250) — matrix operand order repaired and checked by reading, not compiled, not run on hardware
//
#include <hip/hip_runtime.h>
#include <hip/hip_bf16.h>

typedef __attribute__((ext_vector_type(16))) _Float16 v16h;
typedef __attribute__((ext_vector_type(8)))  _Float16 v8h;
typedef __attribute__((ext_vector_type(8)))  float    v8f;
typedef long long i64;

#define N_NODES  50000
#define N_EDGES  800000
#define N_FEAT   128
#define HID      128
#define N_GRAPHS 512

#define CDIV(a, b) (((a) + (b) - 1) / (b))

// Native no-return f32 atomic add at device scope: RMW resolves in L2 (192MB,
// whole working set resident). Avoids any CAS-loop fallback of atomicAdd(float*).
__device__ __forceinline__ void atomic_add_f32_dev(float* p, float v) {
    asm volatile("global_atomic_add_f32 %0, %1, off scope:SCOPE_DEV"
                 :: "v"(p), "v"(v) : "memory");
}

// ---------- elementwise conversions ----------
__global__ void k_f32_to_f16(const float* __restrict__ src, _Float16* __restrict__ dst, int n) {
    int i = blockIdx.x * blockDim.x + threadIdx.x;
    if (i < n) dst[i] = (_Float16)src[i];
}

// W[k*N+n] (in x out, row-major)  ->  Wt[n*K+k]  (out x in) in f16
__global__ void k_transpose_to_f16(const float* __restrict__ W, _Float16* __restrict__ Wt,
                                   int K, int N) {
    int i = blockIdx.x * blockDim.x + threadIdx.x;
    if (i < K * N) {
        int k = i / N, n = i % N;
        Wt[n * K + k] = (_Float16)W[i];
    }
}

// ---------- degree / normalization ----------
__global__ void k_init_deg(float* __restrict__ deg, int n) {
    int i = blockIdx.x * blockDim.x + threadIdx.x;
    if (i < n) deg[i] = 1.0f;   // self-loop contribution
}

__global__ void k_edge_deg(const i64* __restrict__ dst, float* __restrict__ deg, int E) {
    int e = blockIdx.x * blockDim.x + threadIdx.x;
    if (e < E) atomic_add_f32_dev(&deg[(int)dst[e]], 1.0f);
}

__global__ void k_dinv(float* __restrict__ deg, int n) {
    int i = blockIdx.x * blockDim.x + threadIdx.x;
    if (i < n) deg[i] = rsqrtf(deg[i]);  // deg >= 1 always (self-loop)
}

__global__ void k_norm(const i64* __restrict__ src, const i64* __restrict__ dst,
                       const float* __restrict__ dinv, float* __restrict__ norm, int E) {
    int e = blockIdx.x * blockDim.x + threadIdx.x;
    if (e < E) norm[e] = dinv[(int)src[e]] * dinv[(int)dst[e]];
}

// ---------- WMMA GEMM:  C[nRows x Nout] = A[nRows x K] * Bt[Nout x K]^T ----------
// One 16x16 output tile per wave; K-loop of v_wmma_f32_16x16x32_f16.
// A/B use contiguous k = lh*16 + 0..15 in both; C layout per CDNA5 ISA 7.12.2 (wave32).
__global__ void __launch_bounds__(128)
k_wmma_gemm(const _Float16* __restrict__ A, const _Float16* __restrict__ Bt,
            float* __restrict__ C, int nRows, int K, int Nout) {
    int wave = blockIdx.x * (blockDim.x >> 5) + (threadIdx.x >> 5);
    int colTiles = Nout >> 4;
    int nTiles = (nRows >> 4) * colTiles;
    if (wave >= nTiles) return;          // wave-uniform: EXEC stays all-1s for WMMA
    int r = wave / colTiles;
    int c = wave % colTiles;

    int lane = threadIdx.x & 31;
    int lh   = lane >> 4;                // lane-half (0: lanes 0-15, 1: lanes 16-31)
    int l16  = lane & 15;

    // A row for this lane (M = l16), B column for this lane (N = l16)
    const _Float16* arow = A  + (size_t)(r * 16 + l16) * K;
    const _Float16* brow = Bt + (size_t)(c * 16 + l16) * K + lh * 16;

    v8f acc = {};
    for (int kb = 0; kb < K; kb += 32) {
        // A 16x32 f16 fragment: lane holds row M=l16, K = lh*16 + 0..15 (contiguous, matches B)
        v16h a = *(const v16h*)(arow + kb + lh * 16);
        // B 32x16 f16 fragment: lane holds column N=l16, K = lh*16 + 0..15 (contiguous in Bt)
        v16h b = *(const v16h*)(brow + kb);
        acc = __builtin_amdgcn_wmma_f32_16x16x32_f16(
                  false, a, false, b, (short)0, acc, false, false);
    }
    // C f32 16x16: VGPR v holds row M = v + lh*8, col N = l16
    float* crow = C + (size_t)(r * 16 + lh * 8) * Nout + c * 16 + l16;
    #pragma unroll
    for (int v = 0; v < 8; ++v) crow[(size_t)v * Nout] = acc[v];
}

// ---------- edge scatter:  agg[dst] += H[src] * norm  (float4 + native f32 atomics) ----------
template <int F>
__global__ void k_scatter(const float* __restrict__ H, const i64* __restrict__ src,
                          const i64* __restrict__ dst, const float* __restrict__ norm,
                          float* __restrict__ agg, int E) {
    constexpr int CHUNKS = F >> 2;               // float4 chunks per edge (power of 2)
    int tid = blockIdx.x * blockDim.x + threadIdx.x;
    if (tid >= E * CHUNKS) return;
    int e = tid / CHUNKS;                        // pow2 -> shift
    int f = (tid & (CHUNKS - 1)) * 4;
    int s = (int)src[e], d = (int)dst[e];
    float nm = norm[e];
    // prefetch a lookahead edge's source row to hide the random-gather latency
    int ea = e + 16;
    if (ea < E) {
        const float* pf = H + (size_t)((int)src[ea]) * F + f;
        __builtin_prefetch(pf, 0, 0);            // global_prefetch_b8
    }
    const float4 m = *(const float4*)(H + (size_t)s * F + f);
    float* o = agg + (size_t)d * F + f;
    atomic_add_f32_dev(o + 0, m.x * nm);
    atomic_add_f32_dev(o + 1, m.y * nm);
    atomic_add_f32_dev(o + 2, m.z * nm);
    atomic_add_f32_dev(o + 3, m.w * nm);
}

// ---------- finalize: agg + self-loop + bias (+ReLU), emit f16 and/or f32 ----------
__global__ void k_finalize(const float* __restrict__ agg, const float* __restrict__ H,
                           const float* __restrict__ dinv, const float* __restrict__ bias,
                           _Float16* __restrict__ outH, float* __restrict__ outF,
                           int n, int F, int relu) {
    int i = blockIdx.x * blockDim.x + threadIdx.x;
    if (i >= n * F) return;
    int node = i / F, f = i % F;
    float di = dinv[node];
    float v = agg[i] + H[i] * di * di + bias[f];
    if (relu) v = fmaxf(v, 0.0f);
    if (outH) outH[i] = (_Float16)v;
    if (outF) outF[i] = v;
}

// ---------- global mean pool (accumulate) ----------
__global__ void k_pool(const float* __restrict__ G, const i64* __restrict__ batch,
                       float* __restrict__ sums, float* __restrict__ counts, int n) {
    constexpr int F = 64;
    constexpr int CHUNKS = F >> 2;               // 16
    int tid = blockIdx.x * blockDim.x + threadIdx.x;
    if (tid >= n * CHUNKS) return;
    int node = tid / CHUNKS, f = (tid & (CHUNKS - 1)) * 4;
    int g = (int)batch[node];
    const float4 v = *(const float4*)(G + (size_t)node * F + f);
    float* s = sums + (size_t)g * F + f;
    atomic_add_f32_dev(s + 0, v.x);
    atomic_add_f32_dev(s + 1, v.y);
    atomic_add_f32_dev(s + 2, v.z);
    atomic_add_f32_dev(s + 3, v.w);
    if (f == 0) atomic_add_f32_dev(&counts[g], 1.0f);
}

// ---------- head: sigmoid(pooled) @ Wfc + bfc ----------
__global__ void k_head(const float* __restrict__ sums, const float* __restrict__ counts,
                       const float* __restrict__ Wfc, const float* __restrict__ bfc,
                       float* __restrict__ out, int F) {
    int g = blockIdx.x * blockDim.x + threadIdx.x;
    if (g >= N_GRAPHS) return;
    float inv = 1.0f / fmaxf(counts[g], 1.0f);
    float acc = 0.0f;
    for (int f = 0; f < F; ++f) {
        float p = sums[(size_t)g * F + f] * inv;
        float z = 1.0f / (1.0f + expf(-p));
        acc += z * Wfc[f];
    }
    out[g] = acc + bfc[0];
}

extern "C" void kernel_launch(void* const* d_in, const int* in_sizes, int n_in,
                              void* d_out, int out_size, void* d_ws, size_t ws_size,
                              hipStream_t stream) {
    (void)in_sizes; (void)n_in; (void)out_size; (void)ws_size;
    const float* x    = (const float*)d_in[0];
    const i64*   eidx = (const i64*)d_in[1];     // int64 per reference
    const i64*   bat  = (const i64*)d_in[2];
    const float* W1   = (const float*)d_in[3];
    const float* b1   = (const float*)d_in[4];
    const float* W2   = (const float*)d_in[5];
    const float* b2   = (const float*)d_in[6];
    const float* W3   = (const float*)d_in[7];
    const float* b3   = (const float*)d_in[8];
    const float* Wfc  = (const float*)d_in[9];
    const float* bfc  = (const float*)d_in[10];
    float* out = (float*)d_out;

    const i64* esrc = eidx;
    const i64* edst = eidx + N_EDGES;

    // workspace layout (256B aligned slices)
    char* ws = (char*)d_ws;
    size_t off = 0;
    auto alloc = [&](size_t bytes) -> char* {
        char* p = ws + off;
        off += (bytes + 255) & ~(size_t)255;
        return p;
    };
    _Float16* AH  = (_Float16*)alloc(sizeof(_Float16) * N_NODES * HID);  // gemm input (f16)
    _Float16* W1t = (_Float16*)alloc(sizeof(_Float16) * 128 * 128);
    _Float16* W2t = (_Float16*)alloc(sizeof(_Float16) * 64 * 128);
    _Float16* W3t = (_Float16*)alloc(sizeof(_Float16) * 64 * 64);
    float* H    = (float*)alloc(sizeof(float) * N_NODES * HID);          // gemm output
    float* AGG  = (float*)alloc(sizeof(float) * N_NODES * HID);          // scatter accumulator
    float* G    = (float*)alloc(sizeof(float) * N_NODES * 64);           // final node features
    float* DINV = (float*)alloc(sizeof(float) * N_NODES);                // deg -> dinv (in place)
    float* NORM = (float*)alloc(sizeof(float) * N_EDGES);
    float* SUMS = (float*)alloc(sizeof(float) * N_GRAPHS * 64);
    float* CNTS = (float*)alloc(sizeof(float) * N_GRAPHS);

    const int B = 256;

    // 0) x -> f16 ; weights -> transposed f16
    k_f32_to_f16<<<CDIV(N_NODES * N_FEAT, B), B, 0, stream>>>(x, AH, N_NODES * N_FEAT);
    k_transpose_to_f16<<<CDIV(128 * 128, B), B, 0, stream>>>(W1, W1t, 128, 128);
    k_transpose_to_f16<<<CDIV(128 * 64, B), B, 0, stream>>>(W2, W2t, 128, 64);
    k_transpose_to_f16<<<CDIV(64 * 64, B), B, 0, stream>>>(W3, W3t, 64, 64);

    // 1) degree / dinv / per-edge norm
    k_init_deg<<<CDIV(N_NODES, B), B, 0, stream>>>(DINV, N_NODES);
    k_edge_deg<<<CDIV(N_EDGES, B), B, 0, stream>>>(edst, DINV, N_EDGES);
    k_dinv<<<CDIV(N_NODES, B), B, 0, stream>>>(DINV, N_NODES);
    k_norm<<<CDIV(N_EDGES, B), B, 0, stream>>>(esrc, edst, DINV, NORM, N_EDGES);

    // ---- layer 1: 128 -> 128, ReLU ----
    {
        int nTiles = (N_NODES / 16) * (128 / 16);
        k_wmma_gemm<<<CDIV(nTiles, 4), 128, 0, stream>>>(AH, W1t, H, N_NODES, 128, 128);
        hipMemsetAsync(AGG, 0, sizeof(float) * N_NODES * 128, stream);
        k_scatter<128><<<CDIV(N_EDGES * 32, B), B, 0, stream>>>(H, esrc, edst, NORM, AGG, N_EDGES);
        k_finalize<<<CDIV(N_NODES * 128, B), B, 0, stream>>>(AGG, H, DINV, b1, AH, nullptr,
                                                             N_NODES, 128, 1);
    }
    // ---- layer 2: 128 -> 64, ReLU ----
    {
        int nTiles = (N_NODES / 16) * (64 / 16);
        k_wmma_gemm<<<CDIV(nTiles, 4), 128, 0, stream>>>(AH, W2t, H, N_NODES, 128, 64);
        hipMemsetAsync(AGG, 0, sizeof(float) * N_NODES * 64, stream);
        k_scatter<64><<<CDIV(N_EDGES * 16, B), B, 0, stream>>>(H, esrc, edst, NORM, AGG, N_EDGES);
        k_finalize<<<CDIV(N_NODES * 64, B), B, 0, stream>>>(AGG, H, DINV, b2, AH, nullptr,
                                                            N_NODES, 64, 1);
    }
    // ---- layer 3: 64 -> 64, no activation ----
    {
        int nTiles = (N_NODES / 16) * (64 / 16);
        k_wmma_gemm<<<CDIV(nTiles, 4), 128, 0, stream>>>(AH, W3t, H, N_NODES, 64, 64);
        hipMemsetAsync(AGG, 0, sizeof(float) * N_NODES * 64, stream);
        k_scatter<64><<<CDIV(N_EDGES * 16, B), B, 0, stream>>>(H, esrc, edst, NORM, AGG, N_EDGES);
        k_finalize<<<CDIV(N_NODES * 64, B), B, 0, stream>>>(AGG, H, DINV, b3, nullptr, G,
                                                            N_NODES, 64, 0);
    }

    // ---- pool + head ----
    hipMemsetAsync(SUMS, 0, sizeof(float) * N_GRAPHS * 64, stream);
    hipMemsetAsync(CNTS, 0, sizeof(float) * N_GRAPHS, stream);
    k_pool<<<CDIV(N_NODES * 16, B), B, 0, stream>>>(G, bat, SUMS, CNTS, N_NODES);
    k_head<<<CDIV(N_GRAPHS, B), B, 0, stream>>>(SUMS, CNTS, Wfc, bfc, out, 64);
}
